// TritonInt4LinearV2_30855045054754
// MI455X (gfx1250) — compile-verified, matrix operand order checked by reading, not run on hardware
//
#include <hip/hip_runtime.h>
#include <stdint.h>

// ---------------------------------------------------------------------------
// W4A16 grouped-quant GEMM for MI455X (gfx1250, wave32, WMMA).
//   out[8192,11008] f32 = x[8192,4096] f16 @ dequant(int4 W)[4096,11008] + bias
// Compute-bound in fp16 WMMA; dequant uses magic-number nibble->fp16
// (v_lshl_or_b32 + v_and_or_b32 + v_pk_fma_f16, 3 VALU per K-pair) so the
// VALU stream co-executes under the XDL WMMAs. LDS is double-buffered and fed
// by CDNA5 async global->LDS copies; copy addresses are hoisted out of the
// k-loop so each stage issue is just pointer + constant.
// ---------------------------------------------------------------------------

typedef _Float16 __attribute__((ext_vector_type(16))) v16h;
typedef _Float16 __attribute__((ext_vector_type(8)))  v8h;
typedef _Float16 __attribute__((ext_vector_type(2)))  v2h;
typedef float    __attribute__((ext_vector_type(8)))  v8f;

// Types for the async-copy builtin: param0 is
//   'int __attribute__((vector_size(16))) addrspace(1)*' per hipcc diagnostic.
typedef int vsi4 __attribute__((__vector_size__(16)));
typedef __attribute__((address_space(1))) vsi4* as1_v4i;
typedef __attribute__((address_space(3))) vsi4* as3_v4i;

// Problem sizes (fixed by setup_inputs()).
constexpr int kM = 8192;      // B*S = 4*2048
constexpr int kN = 11008;
constexpr int kK = 4096;
constexpr int kG = 128;       // quant group size

// Tiling.
constexpr int kBM = 128;
constexpr int kBN = 128;
constexpr int kBK = 32;                 // one WMMA K-step per block iter
constexpr int kKB = kK / kBK;           // 128 k-blocks
constexpr int kGB = kK / kG;            // 32 scale groups
constexpr int kLDA = kBK + 8;           // 40 halves (80 B row, 16B aligned)
constexpr int kLDW = kBN + 4;           // 132 ints  (528 B row, 16B aligned)

// --------------------- async global->LDS copy (CDNA5) ----------------------
#if defined(__has_builtin)
#  if __has_builtin(__builtin_amdgcn_global_load_async_to_lds_b128)
#    define USE_ASYNC_COPY 1
#  endif
#endif
#ifndef USE_ASYNC_COPY
#  define USE_ASYNC_COPY 0
#endif

__device__ __forceinline__ void copy16(const void* g, void* s) {
#if USE_ASYNC_COPY
  __builtin_amdgcn_global_load_async_to_lds_b128(
      (as1_v4i)g, (as3_v4i)s, 0 /*offset*/, 0 /*cpol*/);
#else
  *(uint4*)s = *(const uint4*)g;
#endif
}

__device__ __forceinline__ void wait_async0() {
#if USE_ASYNC_COPY
#  if defined(__has_builtin) && __has_builtin(__builtin_amdgcn_s_wait_asynccnt)
  __builtin_amdgcn_s_wait_asynccnt(0);
#  else
  asm volatile("s_wait_asynccnt 0" ::: "memory");
#  endif
#endif
}

// ---------------------------------------------------------------------------
__global__ __launch_bounds__(256)
void int4_gemm_wmma(const _Float16* __restrict__ x,
                    const int*      __restrict__ wp,
                    const _Float16* __restrict__ scales,
                    const _Float16* __restrict__ bias,
                    float*          __restrict__ out) {
  __shared__ _Float16 lsA[2][kBM * kLDA];          // 2 * 10240 B
  __shared__ int      lsW[2][(kBK / 2) * kLDW];    // 2 *  8448 B

  const int tid    = threadIdx.x;       // 0..255 (8 waves)
  const int lane   = tid & 31;
  const int wave   = tid >> 5;
  const int waveM  = wave & 1;          // 2-way M split -> 64 rows each
  const int waveN  = wave >> 1;         // 4-way N split -> 32 cols each
  const int r      = lane & 15;
  const int halfId = lane >> 4;         // lane half selects K sub-chunks

  const int mBase = blockIdx.y * kBM;
  const int nBase = blockIdx.x * kBN;

  // ----- per-thread copy coordinates, hoisted out of the k-loop -----------
  // A tile: 128 rows x 64 B = 512 x 16B chunks; thread t takes chunks t, t+256.
  const _Float16* gA   = x + (size_t)(mBase + (tid >> 2)) * kK + (tid & 3) * 8;
  const int       sAo  = (tid >> 2) * kLDA + (tid & 3) * 8;
  // W tile: 16 rows x 512 B = 512 x 16B chunks; thread t takes chunks t, t+256.
  const int*      gW   = wp + (size_t)(tid >> 5) * kN + nBase + (tid & 31) * 4;
  const int       sWo  = (tid >> 5) * kLDW + (tid & 31) * 4;

  auto issue_stage = [&](int kb, int buf) {
    const _Float16* ga = gA + (size_t)kb * kBK;           // advance along K
    copy16(ga,            &lsA[buf][sAo]);
    copy16(ga + 64 * kK,  &lsA[buf][sAo + 64 * kLDA]);    // rows +64
    const int* gw = gW + (size_t)kb * (kBK / 2) * kN;     // advance packed-K
    copy16(gw,            &lsW[buf][sWo]);
    copy16(gw + 8 * kN,   &lsW[buf][sWo + 8 * kLDW]);     // rows +8
  };

  v8f acc[4][2];
#pragma unroll
  for (int mi = 0; mi < 4; ++mi)
#pragma unroll
    for (int ni = 0; ni < 2; ++ni) acc[mi][ni] = v8f{};

  issue_stage(0, 0);

  for (int grp = 0; grp < kGB; ++grp) {
    // Per-lane group scale; magic-FMA constants:
    //   dequant(c) = fma(1024 + c, s, -1032*s) == (c - 8) * s  (single rounding,
    //   but -1032*s is rounded to fp16 first, so not always exact).
    v2h s2[2], bs2[2];
    const v2h m1032 = {(_Float16)(-1032.0f), (_Float16)(-1032.0f)};
#pragma unroll
    for (int ni = 0; ni < 2; ++ni) {
      const _Float16 s = scales[(size_t)grp * kN + nBase + waveN * 32 + ni * 16 + r];
      s2[ni]  = (v2h){s, s};
      bs2[ni] = s2[ni] * m1032;
    }

#pragma unroll
    for (int kbi = 0; kbi < 4; ++kbi) {
      const int kb  = grp * 4 + kbi;
      const int buf = kb & 1;

      wait_async0();          // my stage-kb copies have landed in LDS
      __syncthreads();        // everyone's landed; everyone done with buf^1
      if (kb + 1 < kKB) issue_stage(kb + 1, buf ^ 1);  // overlap with compute

      // A fragments: ISA 16-bit A 16x32 layout.
      // lanes 0-15: K 0..7 and 16..23 ; lanes 16-31: K 8..15 and 24..31.
      const int k0 = halfId ? 8 : 0;
      const int k1 = halfId ? 24 : 16;
      v16h a[4];
#pragma unroll
      for (int mi = 0; mi < 4; ++mi) {
        const _Float16* ab = &lsA[buf][(waveM * 64 + mi * 16 + r) * kLDA];
        v8h lo = *(const v8h*)(ab + k0);   // ds_load_b128
        v8h hi = *(const v8h*)(ab + k1);   // ds_load_b128
        a[mi] = __builtin_shufflevector(lo, hi,
                                        0, 1, 2, 3, 4, 5, 6, 7,
                                        8, 9, 10, 11, 12, 13, 14, 15);
      }

#pragma unroll
      for (int ni = 0; ni < 2; ++ni) {
        // B fragment: VGPR v holds a K-pair; one packed int32 is exactly a
        // K-pair (low nibble = even k, high nibble = odd k). Magic-number
        // dequant: ((w<<12)|w) & 0x000F000F | 0x64006400 -> {1024+lo,1024+hi},
        // then one v_pk_fma_f16 per pair.
        const int col = waveN * 32 + ni * 16 + r;
        union { v16h v; unsigned u[8]; } bf;
#pragma unroll
        for (int v = 0; v < 8; ++v) {
          const int p = (v & 3) + ((v >> 2) << 3) + halfId * 4;  // packed-k row
          const unsigned w = (unsigned)lsW[buf][p * kLDW + col]; // ds_load_b32
          const unsigned t = (((w << 12) | w) & 0x000F000Fu) | 0x64006400u;
          union { unsigned u; v2h h; } cv;
          cv.u = t;
          const v2h d = cv.h * s2[ni] + bs2[ni];                 // v_pk_fma_f16
          union { v2h h; unsigned u; } rs;
          rs.h = d;
          bf.u[v] = rs.u;
        }
#pragma unroll
        for (int mi = 0; mi < 4; ++mi) {
          acc[mi][ni] = __builtin_amdgcn_wmma_f32_16x16x32_f16(
              false, a[mi], false, bf.v, (short)0, acc[mi][ni], false, false);
        }
      }
    }
  }

  // Epilogue: bias add, f32 store. C/D layout: VGPR rr -> M = rr (+8 for
  // lanes 16-31), lane%16 -> N.
  float bfv[2];
#pragma unroll
  for (int ni = 0; ni < 2; ++ni)
    bfv[ni] = (float)bias[nBase + waveN * 32 + ni * 16 + r];

#pragma unroll
  for (int mi = 0; mi < 4; ++mi) {
#pragma unroll
    for (int ni = 0; ni < 2; ++ni) {
      const int col = nBase + waveN * 32 + ni * 16 + r;
#pragma unroll
      for (int rr = 0; rr < 8; ++rr) {
        const int row = mBase + waveM * 64 + mi * 16 + (halfId ? rr + 8 : rr);
        out[(size_t)row * kN + col] = acc[mi][ni][rr] + bfv[ni];
      }
    }
  }
}

// ---------------------------------------------------------------------------
extern "C" void kernel_launch(void* const* d_in, const int* in_sizes, int n_in,
                              void* d_out, int out_size, void* d_ws, size_t ws_size,
                              hipStream_t stream) {
  (void)in_sizes; (void)n_in; (void)out_size; (void)d_ws; (void)ws_size;
  const _Float16* x      = (const _Float16*)d_in[0];
  const int*      wp     = (const int*)d_in[1];
  const _Float16* scales = (const _Float16*)d_in[2];
  const _Float16* bias   = (const _Float16*)d_in[3];
  float*          out    = (float*)d_out;

  dim3 grid(kN / kBN, kM / kBM);   // 86 x 64 blocks
  dim3 block(256);
  hipLaunchKernelGGL(int4_gemm_wmma, grid, block, 0, stream,
                     x, wp, scales, bias, out);
}
